// GATv3Conv_83013127897110
// MI455X (gfx1250) — matrix operand order repaired and checked by reading, not compiled, not run on hardware
//
#include <hip/hip_runtime.h>
#include <hip/hip_bf16.h>

#ifndef __has_builtin
#define __has_builtin(x) 0
#endif

#define N_NODES 50000
#define N_EDGES 800000
#define FEATS   256
#define HEADS   8
#define FPH     32

// LDS A-tile: 16 rows x (256 + 8) halfs -> row stride 528 B (132 dwords; 132 % 64 = 4
// => bank-conflict-free b128 fragment reads)
#define LDS_ROW 264

typedef __attribute__((ext_vector_type(16))) _Float16 v16h;
typedef __attribute__((ext_vector_type(8)))  _Float16 v8h;
typedef __attribute__((ext_vector_type(8)))  float    v8f;
typedef __attribute__((ext_vector_type(4)))  unsigned v4u;
typedef __attribute__((ext_vector_type(8)))  int      v8i;
typedef __attribute__((ext_vector_type(4)))  int      v4i;

__device__ __forceinline__ float silu_f(float x) { return x / (1.0f + __expf(-x)); }

// monotonic float<->uint encoding for atomicMax on signed floats
__device__ __forceinline__ unsigned fenc(float f) {
    unsigned u = __float_as_uint(f);
    return (u & 0x80000000u) ? ~u : (u | 0x80000000u);
}
__device__ __forceinline__ float fdec(unsigned k) {
    unsigned u = (k & 0x80000000u) ? (k & 0x7FFFFFFFu) : ~k;
    return __uint_as_float(u);
}

// ---------------- zero fill ----------------
__global__ __launch_bounds__(256) void zero_u32(unsigned* __restrict__ p, int n) {
    int i = blockIdx.x * 256 + threadIdx.x;
    if (i < n) p[i] = 0u;
}

// ---------------- LayerNorm: one wave32 per row of 256 ----------------
__global__ __launch_bounds__(256) void ln_kernel(const float* __restrict__ feat,
                                                 float* __restrict__ h32,
                                                 _Float16* __restrict__ h16) {
    int row  = blockIdx.x * 8 + (threadIdx.x >> 5);   // 6250 blocks * 8 waves = 50000 rows
    int lane = threadIdx.x & 31;
    const float* p = feat + (size_t)row * FEATS + lane * 8;
    float x[8];
#pragma unroll
    for (int j = 0; j < 8; ++j) x[j] = p[j];
    float s = 0.f, q = 0.f;
#pragma unroll
    for (int j = 0; j < 8; ++j) { s += x[j]; q += x[j] * x[j]; }
#pragma unroll
    for (int off = 16; off > 0; off >>= 1) {
        s += __shfl_xor(s, off, 32);
        q += __shfl_xor(q, off, 32);
    }
    float mu   = s * (1.0f / 256.0f);
    float var  = q * (1.0f / 256.0f) - mu * mu;
    float rstd = rsqrtf(var + 1e-5f);
    float*    o32 = h32 + (size_t)row * FEATS + lane * 8;
    _Float16* o16 = h16 + (size_t)row * FEATS + lane * 8;
#pragma unroll
    for (int j = 0; j < 8; ++j) {
        float v = (x[j] - mu) * rstd;
        o32[j] = v;
        o16[j] = (_Float16)v;
    }
}

// ---------------- f32 -> f16 weight conversion (3 x 256x256) ----------------
__global__ __launch_bounds__(256) void wcvt_kernel(const float* __restrict__ w0,
                                                   const float* __restrict__ w1,
                                                   const float* __restrict__ w2,
                                                   _Float16* __restrict__ w16) {
    int i = blockIdx.x * 256 + threadIdx.x;            // 0 .. 3*65536-1
    int j = i & 65535;
    const float* src = (i < 65536) ? w0 : ((i < 131072) ? w1 : w2);
    w16[i] = (_Float16)src[j];
}

// ---------------- WMMA GEMM with TDM-staged A tile ----------------
// out[N,256] = A[N,256] @ W^T + bias.  One block = one 16-row M-tile.
// The 8 KB A tile is DMA'd to LDS once (Tensor Data Mover, 1-D descriptor with
// pad: 4 dwords every 128 dwords -> 528 B LDS row stride), then 8 waves each
// compute two 16x16 N-tiles (16 x v_wmma_f32_16x16x32_f16 per wave).
__global__ __launch_bounds__(256) void gemm_wmma(const _Float16* __restrict__ A16,
                                                 const _Float16* __restrict__ B16,
                                                 const float* __restrict__ bias,
                                                 float* __restrict__ out) {
    __shared__ _Float16 atile[16 * LDS_ROW];           // 8448 B
    const int tileM = blockIdx.x << 4;
    const int wid   = threadIdx.x >> 5;                // 0..7
    const int lane  = threadIdx.x & 31;
    const int rsel  = lane & 15;
    const int hi    = lane >> 4;                       // 0 or 1

#if __has_builtin(__builtin_amdgcn_tensor_load_to_lds)
    if (threadIdx.x < 32) {                            // one wave issues the TDM op
        unsigned lds_base = (unsigned)(uintptr_t)(&atile[0]);
        unsigned long long ga =
            (unsigned long long)(uintptr_t)(A16 + (size_t)tileM * FEATS);
        v4u g0;
        g0[0] = 1u;                                          // count=1 (valid), user mode
        g0[1] = lds_base;                                    // lds_addr (bytes)
        g0[2] = (unsigned)(ga & 0xFFFFFFFFu);                // global_addr[31:0]
        g0[3] = (unsigned)((ga >> 32) & 0x01FFFFFFu)         // global_addr[56:32]
              | (2u << 30);                                  // type = 2 ("image")
        v8i g1;
        g1[0] = (3 << 16)                                    // data_size = 8 B
              | (1 << 20)                                    // pad_enable
              | (6 << 22)                                    // pad_interval: 128 dwords
              | (3 << 25);                                   // pad_amount: 4 dwords
        g1[1] = (int)(1024u << 16);                          // tensor_dim0 = 1024 (lo16)
        g1[2] = 0;                                           // tensor_dim0 hi / dim1 lo
        g1[3] = (int)(1024u << 16);                          // tile_dim0 = 1024 (8KB 1-D tile)
        g1[4] = 0;                                           // tile_dim1/2 = 0 (unused)
        g1[5] = 1024;                                        // tensor_dim0_stride (unused, 1-D)
        g1[6] = 0;
        g1[7] = 0;
        v4i gz = {0, 0, 0, 0};
#if __clang_major__ >= 23
        v8i gz8 = {0, 0, 0, 0, 0, 0, 0, 0};
        __builtin_amdgcn_tensor_load_to_lds(g0, g1, gz, gz, gz8, 0);
#else
        __builtin_amdgcn_tensor_load_to_lds(g0, g1, gz, gz, 0);
#endif
#if __has_builtin(__builtin_amdgcn_s_wait_tensorcnt)
        __builtin_amdgcn_s_wait_tensorcnt(0);
#else
        asm volatile("s_wait_tensorcnt 0x0" ::: "memory");
#endif
    }
    __syncthreads();
#else
    // cooperative fallback: 512 x 16B chunks with the same padded LDS layout
    {
        const char* gA = (const char*)(A16 + (size_t)tileM * FEATS);
        for (int c = threadIdx.x; c < 512; c += 256) {
            float4 v = *(const float4*)(gA + c * 16);
            *(float4*)((char*)atile + (c >> 5) * (LDS_ROW * 2) + (c & 31) * 16) = v;
        }
    }
    __syncthreads();
#endif

    const int tn0 = (wid * 2) << 4;                    // two N-tiles per wave
    const int tn1 = tn0 + 16;
    const _Float16* b0 = B16 + (size_t)(tn0 + rsel) * FEATS;  // W row n = B column n
    const _Float16* b1 = B16 + (size_t)(tn1 + rsel) * FEATS;
    const _Float16* al = atile + rsel * LDS_ROW;       // A row M = tileM + rsel

    v8f c0 = {}, c1 = {};
#pragma unroll
    for (int k0 = 0; k0 < 256; k0 += 32) {
        // A in this code's contiguous k order (k = 16*hi + s, same as B): lanes 0-15 hold K=0..15; lanes 16-31 K=16..31
        union { v16h v; v8h h[2]; } a;
        a.h[0] = *(const v8h*)(al + k0 + hi * 16);
        a.h[1] = *(const v8h*)(al + k0 + hi * 16 + 8);
        // B in the same contiguous k order as A: lanes 0-15 hold K=0..15 of column n; lanes 16-31 K=16..31
        v16h bb0 = *(const v16h*)(b0 + k0 + hi * 16);
        v16h bb1 = *(const v16h*)(b1 + k0 + hi * 16);
        c0 = __builtin_amdgcn_wmma_f32_16x16x32_f16(false, a.v, false, bb0,
                                                    (short)0, c0, false, false);
        c1 = __builtin_amdgcn_wmma_f32_16x16x32_f16(false, a.v, false, bb1,
                                                    (short)0, c1, false, false);
    }
    // C/D layout: VGPR j -> M = hi*8 + j, N = rsel
    const int orow = tileM + hi * 8;
    float bia0 = bias[tn0 + rsel];
    float bia1 = bias[tn1 + rsel];
    float* po0 = out + (size_t)orow * FEATS + tn0 + rsel;
    float* po1 = out + (size_t)orow * FEATS + tn1 + rsel;
#pragma unroll
    for (int j = 0; j < 8; ++j) {
        po0[(size_t)j * FEATS] = c0[j] + bia0;
        po1[(size_t)j * FEATS] = c1[j] + bia1;
    }
}

// ---------------- edge score: e=silu(fs[u]+fd[v]); score=dot(e,attn); segment max ----------------
__global__ __launch_bounds__(256) void edge_score(const float* __restrict__ fsrc,
                                                  const float* __restrict__ fdst,
                                                  const float* __restrict__ attn,
                                                  const int* __restrict__ src,
                                                  const int* __restrict__ dst,
                                                  float* __restrict__ score,
                                                  unsigned* __restrict__ smax) {
    int gid = blockIdx.x * 256 + threadIdx.x;   // E*H = 6.4M threads
    int e = gid >> 3, hh = gid & 7;
    int s = src[e], d = dst[e];
    const float* ps = fsrc + (size_t)s * FEATS + hh * FPH;
    const float* pd = fdst + (size_t)d * FEATS + hh * FPH;
    const float* pa = attn + hh * FPH;
    float acc = 0.f;
#pragma unroll 8
    for (int f = 0; f < FPH; ++f) {
        float x = ps[f] + pd[f];
        acc += silu_f(x) * pa[f];
    }
    score[gid] = acc;
    atomicMax(&smax[d * 8 + hh], fenc(acc));
}

// ---------------- a = exp(score - max); denom += a ----------------
__global__ __launch_bounds__(256) void edge_norm(const int* __restrict__ dst,
                                                 const unsigned* __restrict__ smax,
                                                 float* __restrict__ score,
                                                 float* __restrict__ denom) {
    int gid = blockIdx.x * 256 + threadIdx.x;   // E*H
    int e = gid >> 3, hh = gid & 7;
    int d = dst[e];
    float m = fdec(smax[d * 8 + hh]);
    float a = __expf(score[gid] - m);
    score[gid] = a;
    atomicAdd(&denom[d * 8 + hh], a);
}

// ---------------- scatter: out[d] += fval[s] * a/denom[d] (float4 per thread) ----------------
__global__ __launch_bounds__(256) void edge_scatter(const float* __restrict__ fval,
                                                    const float* __restrict__ score,
                                                    const float* __restrict__ denom,
                                                    const int* __restrict__ src,
                                                    const int* __restrict__ dst,
                                                    float* __restrict__ out) {
    int gid = blockIdx.x * 256 + threadIdx.x;   // E*64 = 51.2M threads
    int e = gid >> 6;
    int f4 = (gid & 63) << 2;                   // feature base, 0..252
    int hh = f4 >> 5;
    int s = src[e], d = dst[e];
    float a = score[e * 8 + hh] / denom[d * 8 + hh];
    float4 v = *(const float4*)(fval + (size_t)s * FEATS + f4);
    float* po = out + (size_t)d * FEATS + f4;
    atomicAdd(po + 0, v.x * a);
    atomicAdd(po + 1, v.y * a);
    atomicAdd(po + 2, v.z * a);
    atomicAdd(po + 3, v.w * a);
}

// ---------------- residual + silu ----------------
__global__ __launch_bounds__(256) void finish_kernel(float* __restrict__ out,
                                                     const float* __restrict__ h32) {
    int i = blockIdx.x * 256 + threadIdx.x;     // 12.8M
    float v = out[i] + h32[i];
    out[i] = silu_f(v);
}

extern "C" void kernel_launch(void* const* d_in, const int* in_sizes, int n_in,
                              void* d_out, int out_size, void* d_ws, size_t ws_size,
                              hipStream_t stream) {
    const float* feat = (const float*)d_in[0];
    const float* Wsrc = (const float*)d_in[1];
    const float* bsrc = (const float*)d_in[2];
    const float* Wdst = (const float*)d_in[3];
    const float* bdst = (const float*)d_in[4];
    const float* Wval = (const float*)d_in[5];
    const float* bval = (const float*)d_in[6];
    const float* attn = (const float*)d_in[7];
    const int*   src  = (const int*)d_in[8];
    const int*   dst  = (const int*)d_in[9];
    float* out = (float*)d_out;

    // workspace carve-up (all chunks multiple of 512 B)
    char* w = (char*)d_ws;
    float*    h32  = (float*)w;      w += (size_t)N_NODES * FEATS * 4;   // 51.2 MB
    _Float16* h16  = (_Float16*)w;   w += (size_t)N_NODES * FEATS * 2;   // 25.6 MB
    _Float16* w16  = (_Float16*)w;   w += (size_t)3 * 65536 * 2;         // 384 KB
    float*    fsrc = (float*)w;      w += (size_t)N_NODES * FEATS * 4;
    float*    fdst = (float*)w;      w += (size_t)N_NODES * FEATS * 4;
    float*    fval = (float*)w;      w += (size_t)N_NODES * FEATS * 4;
    float*    scor = (float*)w;      w += (size_t)N_EDGES * HEADS * 4;   // 25.6 MB
    unsigned* smax = (unsigned*)w;   w += (size_t)N_NODES * HEADS * 4;
    float*    den  = (float*)w;      w += (size_t)N_NODES * HEADS * 4;

    // zero accumulators (d_out, segment-max keys, denominators)
    zero_u32<<<(N_NODES * FEATS + 255) / 256, 256, 0, stream>>>((unsigned*)out, N_NODES * FEATS);
    zero_u32<<<(N_NODES * HEADS + 255) / 256, 256, 0, stream>>>(smax, N_NODES * HEADS);
    zero_u32<<<(N_NODES * HEADS + 255) / 256, 256, 0, stream>>>((unsigned*)den, N_NODES * HEADS);

    // LayerNorm (f32 + f16 copies)
    ln_kernel<<<N_NODES / 8, 256, 0, stream>>>(feat, h32, h16);

    // weights to f16
    wcvt_kernel<<<(3 * 65536) / 256, 256, 0, stream>>>(Wsrc, Wdst, Wval, w16);

    // three WMMA projections (3125 M-tile blocks each)
    gemm_wmma<<<N_NODES / 16, 256, 0, stream>>>(h16, w16 + 0 * 65536, bsrc, fsrc);
    gemm_wmma<<<N_NODES / 16, 256, 0, stream>>>(h16, w16 + 1 * 65536, bdst, fdst);
    gemm_wmma<<<N_NODES / 16, 256, 0, stream>>>(h16, w16 + 2 * 65536, bval, fval);

    // edge phase
    edge_score<<<(N_EDGES * HEADS) / 256, 256, 0, stream>>>(fsrc, fdst, attn, src, dst, scor, smax);
    edge_norm<<<(N_EDGES * HEADS) / 256, 256, 0, stream>>>(dst, smax, scor, den);
    edge_scatter<<<(N_EDGES * 64) / 256, 256, 0, stream>>>(fval, scor, den, src, dst, out);

    // residual + silu
    finish_kernel<<<(N_NODES * FEATS) / 256, 256, 0, stream>>>(out, h32);
}
